// GPT_76776835384027
// MI455X (gfx1250) — compile-verified, matrix operand order checked by reading, not run on hardware
//
#include <hip/hip_runtime.h>
#include <hip/hip_bf16.h>

typedef __attribute__((ext_vector_type(16))) __bf16 v16bf;
typedef __attribute__((ext_vector_type(8)))  float  v8f;
typedef __attribute__((ext_vector_type(4)))  unsigned u32x4;
typedef __attribute__((ext_vector_type(8)))  int      i32x8;
typedef __attribute__((ext_vector_type(4)))  int      i32x4;

// ---------- helpers ----------
// K index for element j of a 16-bit A/B WMMA fragment (16x32), per lane half.
// half 0: K = {0..7, 16..23}; half 1: K = {8..15, 24..31}  (ISA 7.12.2)
__device__ __forceinline__ int kidx(int j, int hlf) {
  return (j & 7) + ((j >> 3) << 4) + (hlf << 3);
}

__device__ __forceinline__ v8f wmma_bf16(v16bf a, v16bf b, v8f c) {
  return __builtin_amdgcn_wmma_f32_16x16x32_bf16(
      false, a, false, b, (short)0, c, false, false);
}

__device__ __forceinline__ float gelu_tanh(float u) {
  return 0.5f * u * (1.0f + tanhf(0.7978845608028654f * (u + 0.044715f * u * u * u)));
}

// ---------- Tensor Data Mover: 2D fp32 tile Global -> LDS ----------
// D# per CDNA5 ISA:
//  group0: count=1 | lds_addr | global_addr[56:0] | type=2
//  group1: data_size=4B, tensor_dim0/1 (residual extent -> OOB zero-fill),
//          tile_dim0/1, tensor_dim0_stride (row pitch in elements)
__device__ __forceinline__ void tdm_load_2d(unsigned lds_off, const float* gsrc,
                                            unsigned tdim0, unsigned tdim1,
                                            unsigned tile0, unsigned tile1,
                                            unsigned stride0) {
  unsigned long long ga = (unsigned long long)(uintptr_t)gsrc;
  u32x4 g0;
  g0[0] = 1u;                                               // count=1, user desc
  g0[1] = lds_off;                                          // lds_addr (bytes)
  g0[2] = (unsigned)(ga & 0xFFFFFFFFu);                     // global_addr[31:0]
  g0[3] = (unsigned)((ga >> 32) & 0x1FFFFFFu) | (2u << 30); // addr[56:32] | type=2
  i32x8 g1;
  g1[0] = (int)(2u << 16);                                  // wg_mask=0, data_size=2 (4B)
  g1[1] = (int)((tdim0 & 0xFFFFu) << 16);                   // abar=0, tensor_dim0[15:0]
  g1[2] = (int)(((tdim0 >> 16) & 0xFFFFu) | ((tdim1 & 0xFFFFu) << 16));
  g1[3] = (int)(((tdim1 >> 16) & 0xFFFFu) | ((tile0 & 0xFFFFu) << 16));
  g1[4] = (int)(tile1 & 0xFFFFu);                           // tile_dim1, tile_dim2=0
  g1[5] = (int)stride0;                                     // dim0_stride[31:0]
  g1[6] = 0;                                                // dim0_stride[47:32], dim1_stride lo
  g1[7] = 0;                                                // dim1_stride (unused, 2D)
  i32x4 gz = {};
#if defined(__clang_major__) && (__clang_major__ >= 23)
  i32x8 gz8 = {};
  __builtin_amdgcn_tensor_load_to_lds(g0, g1, gz, gz, gz8, 0);
#else
  __builtin_amdgcn_tensor_load_to_lds(g0, g1, gz, gz, 0);
#endif
}

__device__ __forceinline__ unsigned lds_offset(const void* p) {
  return (unsigned)(uintptr_t)p;   // flat LDS addr: low 32 bits = LDS offset
}

// ---------- embedding ----------
__global__ void __launch_bounds__(256) embed_k(const int* __restrict__ tok,
    const float* __restrict__ wte, const float* __restrict__ wpe,
    float* __restrict__ x, int BT, int T, int Cc) {
  int idx = blockIdx.x * blockDim.x + threadIdx.x;
  if (idx >= BT * Cc) return;
  int bt = idx / Cc, c = idx - bt * Cc;
  int t = bt % T;
  int tk = tok[bt];
  x[idx] = wte[(size_t)tk * Cc + c] + wpe[(size_t)t * Cc + c];
}

// ---------- layernorm (one block per row) ----------
__global__ void __launch_bounds__(256) layernorm_k(const float* __restrict__ x,
    const float* __restrict__ g, const float* __restrict__ bta,
    float* __restrict__ o, int Cc) {
  __shared__ float rs[256], rs2[256];
  int row = blockIdx.x;
  const float* xr = x + (size_t)row * Cc;
  float s = 0.f, s2 = 0.f;
  for (int c = threadIdx.x; c < Cc; c += 256) { float v = xr[c]; s += v; s2 += v * v; }
  rs[threadIdx.x] = s; rs2[threadIdx.x] = s2;
  __syncthreads();
  for (int st = 128; st > 0; st >>= 1) {
    if (threadIdx.x < st) { rs[threadIdx.x] += rs[threadIdx.x + st]; rs2[threadIdx.x] += rs2[threadIdx.x + st]; }
    __syncthreads();
  }
  float mean = rs[0] / Cc;
  float var  = rs2[0] / Cc - mean * mean;
  float inv  = rsqrtf(var + 1e-5f);
  float* orow = o + (size_t)row * Cc;
  for (int c = threadIdx.x; c < Cc; c += 256)
    orow[c] = (xr[c] - mean) * inv * g[c] + bta[c];
}

// ---------- bf16 WMMA GEMM: TDM double-buffered staging + LDS bf16 tiles ----------
// flags: bit0 = GELU, bit1 = W transposed (W is [N,K], tied wte head)
#define GFLAG_GELU 1
#define GFLAG_WT   2
__global__ void __launch_bounds__(128) gemm_bf16_k(
    const float* __restrict__ A, const float* __restrict__ W,
    const float* __restrict__ bias, const float* __restrict__ resid,
    float* __restrict__ Cout, int M, int K, int N, int flags) {
  __shared__ float  AsF[2][64 * 32];   // fp32 staging, TDM-filled, double buffered
  __shared__ float  BsF[2][64 * 32];
  __shared__ __bf16 Asb[64 * 32];      // [m][k] bf16, fragment-contiguous
  __shared__ __bf16 Bsb[64 * 32];      // [n][k] bf16, fragment-contiguous
  const int tid = threadIdx.x;
  const int lane = tid & 31;
  const int w = tid >> 5;
  const int hlf = (lane >> 4) & 1;
  const int m0 = blockIdx.y * 64;
  const int n0 = blockIdx.x * 64;
  const int wm = (w >> 1) * 32;
  const int wn = (w & 1) * 32;
  const bool gelu = (flags & GFLAG_GELU) != 0;
  const bool wt   = (flags & GFLAG_WT) != 0;

  auto issue_tiles = [&](int k0, int buf) {
    // A tile: 64 rows (m) x 32 cols (k), row pitch K
    tdm_load_2d(lds_offset(AsF[buf]), A + (size_t)m0 * K + k0,
                (unsigned)(K - k0), (unsigned)(M - m0), 32u, 64u, (unsigned)K);
    // B tile: TDM zero-fills past tensor_dim -> handles the N=50257 edge
    if (wt)   // W[N,K]: 64 rows (n) x 32 cols (k), pitch K -> BsF[n][k]
      tdm_load_2d(lds_offset(BsF[buf]), W + (size_t)n0 * K + k0,
                  (unsigned)(K - k0), (unsigned)(N - n0), 32u, 64u, (unsigned)K);
    else      // W[K,N]: 32 rows (k) x 64 cols (n), pitch N -> BsF[k][n]
      tdm_load_2d(lds_offset(BsF[buf]), W + (size_t)k0 * N + n0,
                  (unsigned)(N - n0), (unsigned)(K - k0), 64u, 32u, (unsigned)N);
  };

  v8f zero = {};
  v8f acc[2][2];
  acc[0][0] = zero; acc[0][1] = zero; acc[1][0] = zero; acc[1][1] = zero;

  const int nIter = K >> 5;
  if (w == 0) issue_tiles(0, 0);

  for (int it = 0; it < nIter; ++it) {
    const int buf = it & 1;
    if (w == 0) __builtin_amdgcn_s_wait_tensorcnt(0);   // tile `it` landed
    __syncthreads();
    if (w == 0 && it + 1 < nIter)                        // DMA next tile ahead
      issue_tiles((it + 1) << 5, (it + 1) & 1);

    // one native cvt per element into fragment-contiguous bf16 tiles
    const float* af32 = AsF[buf];
    const float* bf32 = BsF[buf];
    for (int i = tid; i < 64 * 32; i += 128)
      Asb[i] = (__bf16)af32[i];
    for (int i = tid; i < 64 * 32; i += 128) {
      int n = i >> 5, k = i & 31;
      Bsb[i] = (__bf16)(wt ? bf32[i] : bf32[k * 64 + n]);   // -> [n][k]
    }
    __syncthreads();

    v16bf af[2], bfr[2];
#pragma unroll
    for (int mi = 0; mi < 2; ++mi)
#pragma unroll
      for (int j = 0; j < 16; ++j)
        af[mi][j] = Asb[(wm + mi * 16 + (lane & 15)) * 32 + kidx(j, hlf)];
#pragma unroll
    for (int ni = 0; ni < 2; ++ni)
#pragma unroll
      for (int j = 0; j < 16; ++j)
        bfr[ni][j] = Bsb[(wn + ni * 16 + (lane & 15)) * 32 + kidx(j, hlf)];

#pragma unroll
    for (int mi = 0; mi < 2; ++mi)
#pragma unroll
      for (int ni = 0; ni < 2; ++ni)
        acc[mi][ni] = wmma_bf16(af[mi], bfr[ni], acc[mi][ni]);
    __syncthreads();   // protect Asb/Bsb for next iteration's convert pass
  }

  // epilogue: C layout -> lane = column, VGPR e + 8*half = row
#pragma unroll
  for (int mi = 0; mi < 2; ++mi)
#pragma unroll
    for (int ni = 0; ni < 2; ++ni)
#pragma unroll
      for (int e = 0; e < 8; ++e) {
        int row = m0 + wm + mi * 16 + e + hlf * 8;
        int col = n0 + wn + ni * 16 + (lane & 15);
        if (col < N) {
          float v = acc[mi][ni][e];
          if (bias)  v += bias[col];
          if (gelu)  v = gelu_tanh(v);
          if (resid) v += resid[(size_t)row * N + col];
          Cout[(size_t)row * N + col] = v;
        }
      }
}

// ---------- flash attention: pipelined TDM K/V staging, WMMA QK^T and PV ----------
__global__ void __launch_bounds__(256) attn_k(const float* __restrict__ qkv,
                                              float* __restrict__ y) {
  constexpr int T = 1024, C = 768, HD = 64, NH = 12;
  constexpr float QSCALE = 0.125f;            // 1/sqrt(64)
  __shared__ float  sKf[2][32 * 64];          // fp32 [key][d], TDM, double buffered
  __shared__ float  sVf[2][32 * 64];
  __shared__ __bf16 sKb[32 * 64];             // [key][d] bf16 (S B-fragments)
  __shared__ __bf16 sVb[64 * 32];             // [d][key] bf16 (PV B-fragments)
  __shared__ __bf16 sPb[8][16 * 32];          // per-wave P, [m][k] bf16

  const int tid = threadIdx.x;
  const int lane = tid & 31;
  const int w = tid >> 5;
  const int hlf = (lane >> 4) & 1;
  const int bid = blockIdx.x;
  const int qb = bid % (T / 128);
  const int hh = (bid / (T / 128)) % NH;
  const int b  = bid / ((T / 128) * NH);
  const int q0 = qb * 128 + w * 16;

  auto issue_kv = [&](int kt, int buf) {
    const float* kbase = qkv + (size_t)(b * T + kt) * (3 * C) + C + hh * HD;
    const float* vbase = qkv + (size_t)(b * T + kt) * (3 * C) + 2 * C + hh * HD;
    tdm_load_2d(lds_offset(sKf[buf]), kbase, 64u, (unsigned)(T - kt), 64u, 32u, (unsigned)(3 * C));
    tdm_load_2d(lds_offset(sVf[buf]), vbase, 64u, (unsigned)(T - kt), 64u, 32u, (unsigned)(3 * C));
  };

  // Q fragments (16 queries x 64 dims -> two 16x32 A-fragments), pre-scaled
  v16bf qa[2];
#pragma unroll
  for (int frag = 0; frag < 2; ++frag)
#pragma unroll
    for (int j = 0; j < 16; ++j) {
      int d = kidx(j, hlf) + frag * 32;
      int t = q0 + (lane & 15);
      qa[frag][j] = (__bf16)(qkv[(size_t)(b * T + t) * (3 * C) + hh * HD + d] * QSCALE);
    }

  v8f zero = {};
  v8f o[4]; o[0] = zero; o[1] = zero; o[2] = zero; o[3] = zero;
  float m_i[8], l_i[8];
#pragma unroll
  for (int e = 0; e < 8; ++e) { m_i[e] = -1e30f; l_i[e] = 0.0f; }

  const int nIter = (qb * 128 + 128) >> 5;    // uniform over the block
  if (w == 0) issue_kv(0, 0);

  for (int it = 0; it < nIter; ++it) {
    const int kt = it << 5;
    const int buf = it & 1;
    if (w == 0) __builtin_amdgcn_s_wait_tensorcnt(0);
    __syncthreads();
    if (w == 0 && it + 1 < nIter) issue_kv((it + 1) << 5, (it + 1) & 1);

    // cooperative fp32 -> bf16 convert into fragment-contiguous layouts
    const float* kf = sKf[buf];
    const float* vf = sVf[buf];
    for (int i = tid; i < 32 * 64; i += 256)
      sKb[i] = (__bf16)kf[i];                           // [key][d]
    for (int i = tid; i < 64 * 32; i += 256) {
      int d = i >> 5, key = i & 31;
      sVb[i] = (__bf16)vf[key * 64 + d];                // -> [d][key]
    }
    __syncthreads();

    // S = (Q*scale) @ K^T for two 16-key subtiles
    v8f s[2]; s[0] = zero; s[1] = zero;
#pragma unroll
    for (int sub = 0; sub < 2; ++sub)
#pragma unroll
      for (int frag = 0; frag < 2; ++frag) {
        v16bf kb;
#pragma unroll
        for (int j = 0; j < 16; ++j)
          kb[j] = sKb[(sub * 16 + (lane & 15)) * 64 + kidx(j, hlf) + frag * 32];
        s[sub] = wmma_bf16(qa[frag], kb, s[sub]);
      }

    // causal mask (lane = key column, e + 8*half = query row)
#pragma unroll
    for (int sub = 0; sub < 2; ++sub)
#pragma unroll
      for (int e = 0; e < 8; ++e) {
        int key = kt + sub * 16 + (lane & 15);
        int row = q0 + e + hlf * 8;
        if (key > row) s[sub][e] = -1e30f;
      }

    // online softmax: 16-lane-half row reductions (xor masks stay inside half)
#pragma unroll
    for (int e = 0; e < 8; ++e) {
      float rm = fmaxf(s[0][e], s[1][e]);
      rm = fmaxf(rm, __shfl_xor(rm, 1));
      rm = fmaxf(rm, __shfl_xor(rm, 2));
      rm = fmaxf(rm, __shfl_xor(rm, 4));
      rm = fmaxf(rm, __shfl_xor(rm, 8));
      float mn = fmaxf(m_i[e], rm);
      float alpha = __expf(m_i[e] - mn);
      float p0 = __expf(s[0][e] - mn);
      float p1 = __expf(s[1][e] - mn);
      s[0][e] = p0; s[1][e] = p1;
      float rsum = p0 + p1;
      rsum += __shfl_xor(rsum, 1);
      rsum += __shfl_xor(rsum, 2);
      rsum += __shfl_xor(rsum, 4);
      rsum += __shfl_xor(rsum, 8);
      l_i[e] = l_i[e] * alpha + rsum;
      m_i[e] = mn;
#pragma unroll
      for (int ni = 0; ni < 4; ++ni) o[ni][e] *= alpha;
    }

    // re-layout P (C-layout) -> bf16 A-fragment via per-wave LDS
#pragma unroll
    for (int sub = 0; sub < 2; ++sub)
#pragma unroll
      for (int e = 0; e < 8; ++e)
        sPb[w][(e + hlf * 8) * 32 + sub * 16 + (lane & 15)] = (__bf16)s[sub][e];
    __syncthreads();
    v16bf pa;
#pragma unroll
    for (int j = 0; j < 16; ++j)
      pa[j] = sPb[w][(lane & 15) * 32 + kidx(j, hlf)];

    // O += P @ V  (V chunk 32x64 -> four 32x16 B-fragments)
#pragma unroll
    for (int ni = 0; ni < 4; ++ni) {
      v16bf vb;
#pragma unroll
      for (int j = 0; j < 16; ++j)
        vb[j] = sVb[(ni * 16 + (lane & 15)) * 32 + kidx(j, hlf)];
      o[ni] = wmma_bf16(pa, vb, o[ni]);
    }
    __syncthreads();   // protect sKb/sVb/sPb for next iteration
  }

  // normalize and store
#pragma unroll
  for (int e = 0; e < 8; ++e) {
    float inv = 1.0f / l_i[e];
    int row = q0 + e + hlf * 8;
#pragma unroll
    for (int ni = 0; ni < 4; ++ni)
      y[(size_t)(b * T + row) * C + hh * HD + ni * 16 + (lane & 15)] = o[ni][e] * inv;
  }
}

// ---------- loss ----------
__global__ void __launch_bounds__(256) loss_row_k(const float* __restrict__ logits,
    const int* __restrict__ target, float* __restrict__ nll, int Vv) {
  __shared__ float red[256];
  int row = blockIdx.x;
  const float* lr = logits + (size_t)row * Vv;
  float mx = -1e30f;
  for (int i = threadIdx.x; i < Vv; i += 256) mx = fmaxf(mx, lr[i]);
  red[threadIdx.x] = mx; __syncthreads();
  for (int st = 128; st > 0; st >>= 1) {
    if (threadIdx.x < st) red[threadIdx.x] = fmaxf(red[threadIdx.x], red[threadIdx.x + st]);
    __syncthreads();
  }
  mx = red[0];
  __syncthreads();
  float s = 0.f;
  for (int i = threadIdx.x; i < Vv; i += 256) s += __expf(lr[i] - mx);
  red[threadIdx.x] = s; __syncthreads();
  for (int st = 128; st > 0; st >>= 1) {
    if (threadIdx.x < st) red[threadIdx.x] += red[threadIdx.x + st];
    __syncthreads();
  }
  if (threadIdx.x == 0) {
    float lse = logf(red[0]) + mx;
    nll[row] = lse - lr[target[row]];
  }
}

__global__ void __launch_bounds__(256) loss_mean_k(const float* __restrict__ nll,
                                                   int n, float* __restrict__ out) {
  __shared__ float red[256];
  float s = 0.f;
  for (int i = threadIdx.x; i < n; i += 256) s += nll[i];
  red[threadIdx.x] = s; __syncthreads();
  for (int st = 128; st > 0; st >>= 1) {
    if (threadIdx.x < st) red[threadIdx.x] += red[threadIdx.x + st];
    __syncthreads();
  }
  if (threadIdx.x == 0) out[0] = red[0] / (float)n;
}

// ---------- host orchestration ----------
extern "C" void kernel_launch(void* const* d_in, const int* in_sizes, int n_in,
                              void* d_out, int out_size, void* d_ws, size_t ws_size,
                              hipStream_t stream) {
  constexpr int B = 4, T = 1024, Cc = 768, NHn = 12, NLn = 12, Vv = 50257;
  const int BT = B * T;

  const int*   tokens = (const int*)d_in[0];
  const int*   target = (const int*)d_in[1];
  const float* wte    = (const float*)d_in[2];
  const float* wpe    = (const float*)d_in[3];
  const float* ln1_g  = (const float*)d_in[4];
  const float* ln1_b  = (const float*)d_in[5];
  const float* attn_w = (const float*)d_in[6];
  const float* attn_b = (const float*)d_in[7];
  const float* proj_w = (const float*)d_in[8];
  const float* proj_b = (const float*)d_in[9];
  const float* ln2_g  = (const float*)d_in[10];
  const float* ln2_b  = (const float*)d_in[11];
  const float* fc_w   = (const float*)d_in[12];
  const float* fc_b   = (const float*)d_in[13];
  const float* fc2_w  = (const float*)d_in[14];
  const float* fc2_b  = (const float*)d_in[15];
  const float* lnf_g  = (const float*)d_in[16];
  const float* lnf_b  = (const float*)d_in[17];

  float* ws  = (float*)d_ws;
  float* x   = ws;
  float* h   = x   + (size_t)BT * Cc;
  float* qkv = h   + (size_t)BT * Cc;
  float* y   = qkv + (size_t)BT * 3 * Cc;
  float* fc  = y   + (size_t)BT * Cc;
  float* nll = fc  + (size_t)BT * 4 * Cc;

  embed_k<<<(BT * Cc + 255) / 256, 256, 0, stream>>>(tokens, wte, wpe, x, BT, T, Cc);

  for (int l = 0; l < NLn; ++l) {
    layernorm_k<<<BT, 256, 0, stream>>>(x, ln1_g + l * Cc, ln1_b + l * Cc, h, Cc);
    gemm_bf16_k<<<dim3(3 * Cc / 64, BT / 64), 128, 0, stream>>>(
        h, attn_w + (size_t)l * Cc * 3 * Cc, attn_b + (size_t)l * 3 * Cc,
        nullptr, qkv, BT, Cc, 3 * Cc, 0);
    attn_k<<<B * NHn * (T / 128), 256, 0, stream>>>(qkv, y);
    gemm_bf16_k<<<dim3(Cc / 64, BT / 64), 128, 0, stream>>>(
        y, proj_w + (size_t)l * Cc * Cc, proj_b + (size_t)l * Cc,
        x, x, BT, Cc, Cc, 0);
    layernorm_k<<<BT, 256, 0, stream>>>(x, ln2_g + l * Cc, ln2_b + l * Cc, h, Cc);
    gemm_bf16_k<<<dim3(4 * Cc / 64, BT / 64), 128, 0, stream>>>(
        h, fc_w + (size_t)l * Cc * 4 * Cc, fc_b + (size_t)l * 4 * Cc,
        nullptr, fc, BT, Cc, 4 * Cc, GFLAG_GELU);
    gemm_bf16_k<<<dim3(Cc / 64, BT / 64), 128, 0, stream>>>(
        fc, fc2_w + (size_t)l * 4 * Cc * Cc, fc2_b + (size_t)l * Cc,
        x, x, BT, 4 * Cc, Cc, 0);
  }

  layernorm_k<<<BT, 256, 0, stream>>>(x, lnf_g, lnf_b, h, Cc);

  float* logits = (float*)d_out;
  gemm_bf16_k<<<dim3((Vv + 63) / 64, BT / 64), 128, 0, stream>>>(
      h, wte, nullptr, nullptr, logits, BT, Cc, Vv, GFLAG_WT);

  loss_row_k<<<BT, 256, 0, stream>>>(logits, target, nll, Vv);
  loss_mean_k<<<1, 256, 0, stream>>>(nll, BT, logits + (size_t)BT * Vv);
}
